// HeriarchicalClassedProjectedAdaptiveLogSoftmax_2199023256017
// MI455X (gfx1250) — matrix operand order repaired and checked by reading, not compiled, not run on hardware
//
#include <hip/hip_runtime.h>

// ---------------------------------------------------------------------------
// Adaptive log-softmax NLL for MI455X (gfx1250, wave32, WMMA).
//
// Roofline: dense logits = 560 GFLOP vs 1.07GB W traffic (46us @23.3TB/s)
// -> compute bound -> bf16 WMMA (v_wmma_f32_16x16x32_bf16), fp32 accum.
// W is read from HBM exactly once: each block pins its full 128-col x 1024-K
// bf16 B-tile (256KB) in the 320KB LDS and sweeps all 1024 hidden rows.
// A-tiles are staged with GLOBAL_LOAD_ASYNC_TO_LDS_B128 (ASYNCcnt) so the
// global latency is hidden behind the WMMA work of the current K-chunk.
// Softmax (max, sum-exp) is fused into the GEMM epilogue per 128x128 tile.
// ---------------------------------------------------------------------------

typedef __attribute__((ext_vector_type(16))) __bf16          v16bf;
typedef __attribute__((ext_vector_type(8)))  __bf16          v8bf;
typedef __attribute__((ext_vector_type(8)))  float           v8f;
typedef __attribute__((ext_vector_type(8)))  unsigned short  v8us;

#define NEG_HUGE (-3.0e38f)

static constexpr int kD    = 1024;   // hidden dim
static constexpr int kRows = 1024;   // tokens
static constexpr int BN    = 128;    // column tile

// Segment-aligned tile table (virtual column space):
//  seg0 head   : 20002 cols (W[0:20000] + 2 cluster rows) -> tiles [0,157)
//  seg1        : W[20000:20008)  -> tile  [157,158)
//  seg2        : W[20008:20016)  -> tile  [158,159)
//  seg3        : W[20016:200000) -> tiles [159,1566)
//  seg4        : W[200000:267735)-> tiles [1566,2096)
static constexpr int T0_END = 157;
static constexpr int T1_END = 158;
static constexpr int T2_END = 159;
static constexpr int T3_END = 1566;
static constexpr int T4_END = 2096;

// fp32 -> bf16, round-to-nearest-even (integer domain: no __bf16 converts needed)
__device__ __forceinline__ unsigned short f2bf(float f) {
  union { float f; unsigned int u; } x; x.f = f;
  unsigned int u = x.u;
  unsigned int r = u + 0x7FFFu + ((u >> 16) & 1u);
  return (unsigned short)(r >> 16);
}

__device__ __forceinline__ void lse_merge(float& m, float& s, float m2, float s2) {
  float M = fmaxf(m, m2);
  s = s * __expf(m - M) + s2 * __expf(m2 - M);
  m = M;
}

// Async global->LDS copy of 32 bytes (two B128 ops), tracked by ASYNCcnt.
// Low 32 bits of a flat shared pointer are the wave-relative LDS offset.
__device__ __forceinline__ void async_copy32(unsigned short* ldst,
                                             const unsigned short* gsrc) {
  unsigned int lds_off = (unsigned int)(uintptr_t)ldst;
  unsigned long long gaddr = (unsigned long long)(uintptr_t)gsrc;
  asm volatile("global_load_async_to_lds_b128 %0, %1, off"
               :: "v"(lds_off), "v"(gaddr) : "memory");
  asm volatile("global_load_async_to_lds_b128 %0, %1, off offset:16"
               :: "v"(lds_off), "v"(gaddr) : "memory");
}

__device__ __forceinline__ void wait_async0() {
#if __has_builtin(__builtin_amdgcn_s_wait_asynccnt)
  __builtin_amdgcn_s_wait_asynccnt(0);
#else
  asm volatile("s_wait_asynccnt 0x0" ::: "memory");
#endif
}

// ---------------------------------------------------------------------------
// Stage 0a: hidden fp32 -> bf16 (row-major [1024][1024]) in workspace
// ---------------------------------------------------------------------------
__global__ void cvt_hidden_kernel(const float* __restrict__ h,
                                  unsigned short* __restrict__ hb) {
  int i = (blockIdx.x * blockDim.x + threadIdx.x) * 8;
  float4 f0 = *(const float4*)(h + i);
  float4 f1 = *(const float4*)(h + i + 4);
  v8us o;
  o[0] = f2bf(f0.x); o[1] = f2bf(f0.y); o[2] = f2bf(f0.z); o[3] = f2bf(f0.w);
  o[4] = f2bf(f1.x); o[5] = f2bf(f1.y); o[6] = f2bf(f1.z); o[7] = f2bf(f1.w);
  *(v8us*)(hb + i) = o;
}

// ---------------------------------------------------------------------------
// Stage 0b: per-row routing. Each row needs at most 2 specific logits:
//   pick0 = head logit (target col if t<20000, else routing col 0/1/20001/20000)
//   pick1 = tail target logit (virtual column), or -1 if none.
// ---------------------------------------------------------------------------
__global__ void make_picks_kernel(const int* __restrict__ target,
                                  int* __restrict__ p0, int* __restrict__ p1,
                                  int* __restrict__ seg) {
  int n = blockIdx.x * blockDim.x + threadIdx.x;
  if (n >= kRows) return;
  int t = target[n];
  int a, bb = -1, sg = 0;
  if (t < 20000)       { a = t; }
  else if (t < 20008)  { a = 0;     bb = 20002  + (t - 20000);  sg = 1; }
  else if (t < 20016)  { a = 1;     bb = 20010  + (t - 20008);  sg = 2; }
  else if (t < 200000) { a = 20001; bb = 20018  + (t - 20016);  sg = 3; }
  else                 { a = 20000; bb = 200002 + (t - 200000); sg = 4; }
  p0[n] = a; p1[n] = bb; seg[n] = sg;
}

// ---------------------------------------------------------------------------
// Stage A: per column-tile: convert W tile -> bf16 LDS (once), then for each
// of 8 M-chunks run the K=1024 WMMA loop and a fused logsumexp epilogue.
// Block: 256 threads = 8 waves in a 2(M) x 4(N) grid; wave owns 64x32 output.
// ---------------------------------------------------------------------------
__global__ void __launch_bounds__(256)
adaptive_logits_kernel(const float* __restrict__ W,
                       const float* __restrict__ b,
                       const float* __restrict__ cw,
                       const float* __restrict__ cb,
                       const unsigned short* __restrict__ hb,
                       const int* __restrict__ p0v,
                       const int* __restrict__ p1v,
                       float* __restrict__ pv0,
                       float* __restrict__ pv1,
                       float2* __restrict__ partial) {
  extern __shared__ char smem[];
  unsigned short* Bs   = (unsigned short*)smem;                          // 128*1024*2 = 262144
  unsigned short* As   = (unsigned short*)(smem + 262144);               // 2*128*32*2 =  16384
  float*          biasS= (float*)(smem + 262144 + 16384);                // 128*4      =    512
  float2*         redS = (float2*)(smem + 262144 + 16384 + 512);         // 8*64*8     =   4096
  int*            p0S  = (int*)(smem + 262144 + 16384 + 512 + 4096);     // 1024*4
  int*            p1S  = (int*)(smem + 262144 + 16384 + 512 + 4096 + 4096);

  const int tid  = threadIdx.x;
  const int tile = blockIdx.x;

  int seg, colInSeg0, segCols, wbase, vcol0;
  if (tile < T0_END)      { seg = 0; colInSeg0 = tile * BN;            segCols = 20002;  wbase = 0;      vcol0 = colInSeg0;          }
  else if (tile < T1_END) { seg = 1; colInSeg0 = (tile - T0_END) * BN; segCols = 8;      wbase = 20000;  vcol0 = 20002  + colInSeg0; }
  else if (tile < T2_END) { seg = 2; colInSeg0 = (tile - T1_END) * BN; segCols = 8;      wbase = 20008;  vcol0 = 20010  + colInSeg0; }
  else if (tile < T3_END) { seg = 3; colInSeg0 = (tile - T2_END) * BN; segCols = 179984; wbase = 20016;  vcol0 = 20018  + colInSeg0; }
  else                    { seg = 4; colInSeg0 = (tile - T3_END) * BN; segCols = 67735;  wbase = 200000; vcol0 = 200002 + colInSeg0; }

  // ---- Load B tile into LDS: 128 cols x 1024 K, fp32 -> bf16 (column = W row)
  {
    int c  = tid >> 1;   // 0..127 column in tile
    int kh = tid & 1;    // K half (512 each)
    int colInSeg = colInSeg0 + c;
    bool colValid = colInSeg < segCols;
    const float* src = nullptr;
    float bias_v = 0.f;
    if (colValid) {
      if (seg == 0) {
        if (colInSeg < 20000) { src = W  + (size_t)colInSeg * kD;          bias_v = b[colInSeg]; }
        else                  { src = cw + (size_t)(colInSeg - 20000) * kD; bias_v = cb[colInSeg - 20000]; }
      } else {
        int wr = wbase + colInSeg;
        src = W + (size_t)wr * kD; bias_v = b[wr];
      }
    }
    if (kh == 0) biasS[c] = bias_v;
    unsigned short* dst = Bs + (size_t)c * kD + kh * 512;
    if (src) {
      src += kh * 512;
      for (int k = 0; k < 512; k += 8) {
        __builtin_prefetch(src + k + 256, 0, 1);   // global_prefetch_b8 stream hint
        float4 f0 = *(const float4*)(src + k);
        float4 f1 = *(const float4*)(src + k + 4);
        v8us o;
        o[0] = f2bf(f0.x); o[1] = f2bf(f0.y); o[2] = f2bf(f0.z); o[3] = f2bf(f0.w);
        o[4] = f2bf(f1.x); o[5] = f2bf(f1.y); o[6] = f2bf(f1.z); o[7] = f2bf(f1.w);
        *(v8us*)(dst + k) = o;
      }
    } else {
      v8us z; for (int e = 0; e < 8; ++e) z[e] = 0;
      for (int k = 0; k < 512; k += 8) *(v8us*)(dst + k) = z;
    }
    for (int r = tid; r < kRows; r += 256) { p0S[r] = p0v[r]; p1S[r] = p1v[r]; }
  }
  __syncthreads();

  const int lane  = tid & 31;
  const int wid   = tid >> 5;
  const int waveM = wid >> 2;          // 0..1 -> 64 rows each
  const int waveN = wid & 3;           // 0..3 -> 32 cols each
  const int halfL = lane >> 4;
  const int ln16  = lane & 15;
  int validCols = segCols - colInSeg0; if (validCols > BN) validCols = BN;

  const int arow  = tid >> 1;          // A-staging: 2 threads per row, 32B each
  const int apart = tid & 1;

  for (int mc = 0; mc < 8; ++mc) {
    const int mbase = mc * 128;
    v8f acc[4][2];
    #pragma unroll
    for (int i = 0; i < 4; ++i)
      #pragma unroll
      for (int j = 0; j < 2; ++j)
        #pragma unroll
        for (int k = 0; k < 8; ++k) acc[i][j][k] = 0.f;

    // prime A double-buffer (buffer 0, kb=0) via async global->LDS
    async_copy32(As + arow * 32 + apart * 16,
                 hb + (size_t)(mbase + arow) * kD + apart * 16);
    wait_async0();
    __syncthreads();

    for (int kb = 0; kb < kD; kb += 32) {
      const int cur = (kb >> 5) & 1;
      if (kb + 32 < kD) {   // async-stage next A chunk; latency hidden by WMMAs
        async_copy32(As + (cur ^ 1) * (128 * 32) + arow * 32 + apart * 16,
                     hb + (size_t)(mbase + arow) * kD + (kb + 32) + apart * 16);
      }
      const unsigned short* Abuf = As + cur * (128 * 32);

      // B fragments: 32x16 bf16, lane holds col N=ln16, K interleaved by half (8*half+e, 16+8*half+e)
      v16bf bfrag[2];
      #pragma unroll
      for (int j = 0; j < 2; ++j) {
        const unsigned short* colbase = Bs + (size_t)(waveN * 32 + j * 16 + ln16) * kD + kb;
        const v8bf b0 = *(const v8bf*)(colbase + halfL * 8);
        const v8bf b1 = *(const v8bf*)(colbase + 16 + halfL * 8);
        #pragma unroll
        for (int e = 0; e < 8; ++e) { bfrag[j][e] = b0[e]; bfrag[j][8 + e] = b1[e]; }
      }
      // A fragments: 16x32 bf16, lane holds row M=ln16, K interleaved by half
      #pragma unroll
      for (int i = 0; i < 4; ++i) {
        const unsigned short* rowbase = Abuf + (waveM * 64 + i * 16 + ln16) * 32;
        const v8bf a0 = *(const v8bf*)(rowbase + halfL * 8);
        const v8bf a1 = *(const v8bf*)(rowbase + 16 + halfL * 8);
        v16bf afrag;
        #pragma unroll
        for (int e = 0; e < 8; ++e) { afrag[e] = a0[e]; afrag[8 + e] = a1[e]; }
        #pragma unroll
        for (int j = 0; j < 2; ++j)
          acc[i][j] = __builtin_amdgcn_wmma_f32_16x16x32_bf16(
              false, afrag, false, bfrag[j], (short)0, acc[i][j], false, false);
      }
      wait_async0();       // staged LDS chunk complete before the barrier
      __syncthreads();
    }

    // ---- fused epilogue: bias add, pick capture, per-row (max, sum-exp) ----
    #pragma unroll
    for (int i = 0; i < 4; ++i) {
      #pragma unroll
      for (int k = 0; k < 8; ++k) {
        const int rowL  = i * 16 + k + 8 * halfL;       // 0..63 within wave
        const int row_g = mbase + waveM * 64 + rowL;
        const int c0 = waveN * 32 + ln16;
        const int c1 = c0 + 16;
        const float v0 = acc[i][0][k] + biasS[c0];
        const float v1 = acc[i][1][k] + biasS[c1];
        const bool a0 = c0 < validCols;
        const bool a1 = c1 < validCols;
        const int q0 = p0S[row_g], q1 = p1S[row_g];
        const int vg0 = vcol0 + c0, vg1 = vcol0 + c1;
        if (a0) { if (vg0 == q0) pv0[row_g] = v0; if (vg0 == q1) pv1[row_g] = v0; }
        if (a1) { if (vg1 == q0) pv0[row_g] = v1; if (vg1 == q1) pv1[row_g] = v1; }
        float m = fmaxf(a0 ? v0 : NEG_HUGE, a1 ? v1 : NEG_HUGE);
        float s = (a0 ? __expf(v0 - m) : 0.f) + (a1 ? __expf(v1 - m) : 0.f);
        #pragma unroll
        for (int d = 1; d < 16; d <<= 1) {   // reduce across the 16-lane row group
          float mo = __shfl_xor(m, d, 16);
          float so = __shfl_xor(s, d, 16);
          lse_merge(m, s, mo, so);
        }
        if (ln16 == 0) redS[wid * 64 + rowL] = make_float2(m, s);
      }
    }
    __syncthreads();
    if (tid < 128) {   // merge the 4 N-waves per row, emit tile partial
      const int wm = tid >> 6, rl = tid & 63;
      float m = NEG_HUGE, s = 0.f;
      #pragma unroll
      for (int wn = 0; wn < 4; ++wn) {
        float2 p = redS[(wm * 4 + wn) * 64 + rl];
        lse_merge(m, s, p.x, p.y);
      }
      partial[(size_t)tile * kRows + (mbase + wm * 64 + rl)] = make_float2(m, s);
    }
    __syncthreads();
  }
}

// ---------------------------------------------------------------------------
// Stage B: per row, merge tile partials (head + its tail segment) -> NLL.
// ---------------------------------------------------------------------------
__global__ void finalize_kernel(const float2* __restrict__ partial,
                                const int* __restrict__ segid,
                                const float* __restrict__ pv0,
                                const float* __restrict__ pv1,
                                float* __restrict__ out) {
  __shared__ float2 sh[4];
  __shared__ float2 res;
  const int n = blockIdx.x;
  const int tid = threadIdx.x;   // 128 threads

  // head segment: tiles [0, T0_END)
  float m = NEG_HUGE, s = 0.f;
  for (int t = tid; t < T0_END; t += 128) {
    float2 p = partial[(size_t)t * kRows + n];
    lse_merge(m, s, p.x, p.y);
  }
  #pragma unroll
  for (int d = 1; d < 32; d <<= 1) {
    float mo = __shfl_xor(m, d, 32);
    float so = __shfl_xor(s, d, 32);
    lse_merge(m, s, mo, so);
  }
  if ((tid & 31) == 0) sh[tid >> 5] = make_float2(m, s);
  __syncthreads();
  if (tid == 0) {
    float M = sh[0].x, S = sh[0].y;
    for (int w = 1; w < 4; ++w) lse_merge(M, S, sh[w].x, sh[w].y);
    res = make_float2(M, S);
  }
  __syncthreads();
  const float2 H = res;
  __syncthreads();

  const int sg = segid[n];            // uniform across block
  float2 T = make_float2(0.f, 1.f);
  if (sg > 0) {
    int ts, te;
    if (sg == 1)      { ts = T0_END; te = T1_END; }
    else if (sg == 2) { ts = T1_END; te = T2_END; }
    else if (sg == 3) { ts = T2_END; te = T3_END; }
    else              { ts = T3_END; te = T4_END; }
    m = NEG_HUGE; s = 0.f;
    for (int t = ts + tid; t < te; t += 128) {
      float2 p = partial[(size_t)t * kRows + n];
      lse_merge(m, s, p.x, p.y);
    }
    #pragma unroll
    for (int d = 1; d < 32; d <<= 1) {
      float mo = __shfl_xor(m, d, 32);
      float so = __shfl_xor(s, d, 32);
      lse_merge(m, s, mo, so);
    }
    if ((tid & 31) == 0) sh[tid >> 5] = make_float2(m, s);
    __syncthreads();
    if (tid == 0) {
      float M = sh[0].x, S = sh[0].y;
      for (int w = 1; w < 4; ++w) lse_merge(M, S, sh[w].x, sh[w].y);
      res = make_float2(M, S);
    }
    __syncthreads();
    T = res;
  }

  if (tid == 0) {
    float r = (H.x + __logf(H.y)) - pv0[n];          // head_lse - head/route logit
    if (sg > 0) r += (T.x + __logf(T.y)) - pv1[n];   // + tail_lse - tail logit
    out[n] = r;
  }
}

// ---------------------------------------------------------------------------
// Host launcher. Workspace layout (bytes):
//   [0)        hidden bf16          2,097,152
//   [2097152)  pick0 vcol (int)         4,096
//   [2101248)  pick1 vcol (int)         4,096
//   [2105344)  seg id (int)             4,096
//   [2109440)  pick0 val (float)        4,096
//   [2113536)  pick1 val (float)        4,096
//   [2117632)  tile partials float2 17,170,432   (2096 tiles x 1024 rows)
// ---------------------------------------------------------------------------
extern "C" void kernel_launch(void* const* d_in, const int* in_sizes, int n_in,
                              void* d_out, int out_size, void* d_ws, size_t ws_size,
                              hipStream_t stream) {
  const float* hidden = (const float*)d_in[0];
  const int*   target = (const int*)d_in[1];
  const float* W      = (const float*)d_in[2];
  const float* b      = (const float*)d_in[3];
  const float* cw     = (const float*)d_in[4];
  const float* cb     = (const float*)d_in[5];
  float* out = (float*)d_out;
  char*  ws  = (char*)d_ws;

  unsigned short* hb  = (unsigned short*)(ws + 0);
  int*    p0  = (int*)(ws + 2097152);
  int*    p1  = (int*)(ws + 2101248);
  int*    sg  = (int*)(ws + 2105344);
  float*  pv0 = (float*)(ws + 2109440);
  float*  pv1 = (float*)(ws + 2113536);
  float2* part= (float2*)(ws + 2117632);

  cvt_hidden_kernel<<<512, 256, 0, stream>>>(hidden, hb);   // 1M elems, 8/thread
  make_picks_kernel<<<4, 256, 0, stream>>>(target, p0, p1, sg);

  const size_t smem = 262144 + 16384 + 512 + 4096 + 4096 + 4096;  // 291,328 B < 320KB
  adaptive_logits_kernel<<<T4_END, 256, smem, stream>>>(W, b, cw, cb, hb,
                                                        p0, p1, pv0, pv1, part);
  finalize_kernel<<<kRows, 128, 0, stream>>>(part, sg, pv0, pv1, out);
}
